// MultiHeadAttention_4063039062493
// MI455X (gfx1250) — matrix operand order repaired and checked by reading, not compiled, not run on hardware
//
#include <hip/hip_runtime.h>
#include <hip/hip_bf16.h>

// ---------------------------------------------------------------------------
// Fused causal multi-head attention forward, bf16 WMMA + async global->LDS
// staging (gfx1250 / MI455X).  B=2, T=2048, C=1024, H=16, D=64, N3=3C=3072
// ---------------------------------------------------------------------------

typedef __attribute__((ext_vector_type(16))) __bf16 v16bf;
typedef __attribute__((ext_vector_type(8)))  __bf16 v8bf;
typedef __attribute__((ext_vector_type(4)))  __bf16 v4bf;
typedef __attribute__((ext_vector_type(8)))  float  v8f;
typedef __attribute__((ext_vector_type(4)))  float  v4f;

#define T_SEQ   2048
#define C_EMB   1024
#define N_HEADS 16
#define D_HEAD  64
#define N3      3072

__device__ __forceinline__ __bf16 f2bf(float f) {
  unsigned u = __float_as_uint(f);
  unsigned r = u + 0x7FFFu + ((u >> 16) & 1u);   // round-to-nearest-even
  unsigned short h = (unsigned short)(r >> 16);
  return __builtin_bit_cast(__bf16, h);
}

__device__ __forceinline__ v16bf cat8(v8bf lo, v8bf hi) {
  return __builtin_shufflevector(lo, hi, 0,1,2,3,4,5,6,7,8,9,10,11,12,13,14,15);
}

// Async DMA: 16 bytes global -> LDS, no VGPR data, tracked by ASYNCcnt.
__device__ __forceinline__ void async_cp16(unsigned lds_off, const void* gaddr) {
  asm volatile("global_load_async_to_lds_b128 %0, %1, off"
               :: "v"(lds_off), "v"(gaddr) : "memory");
}

// ---------------------------------------------------------------------------
// f32 -> bf16 conversion (vectorized x4)
// ---------------------------------------------------------------------------
__global__ void __launch_bounds__(256) cvt_f32_bf16(const v4f* __restrict__ in,
                                                    v4bf* __restrict__ out, int n4) {
  int i = blockIdx.x * blockDim.x + threadIdx.x;
  if (i < n4) {
    v4f x = in[i];
    v4bf y;
    y[0] = f2bf(x[0]); y[1] = f2bf(x[1]); y[2] = f2bf(x[2]); y[3] = f2bf(x[3]);
    out[i] = y;
  }
}

// ---------------------------------------------------------------------------
// QKV projection: [4096 x 1024] x [1024 x 3072] + bias.
// 4 waves / block; wave computes a 16x64 strip via v_wmma_f32_16x16x32_bf16.
// Writes Q [B,H,T,D] (pre-scaled by log2(e)/sqrt(D)), Kt [B,H,D,T], V [B,H,T,D].
// ---------------------------------------------------------------------------
__global__ void __launch_bounds__(128) qkv_gemm(
    const __bf16* __restrict__ xb, const __bf16* __restrict__ wb,
    const float* __restrict__ bias,
    __bf16* __restrict__ Q, __bf16* __restrict__ Kt, __bf16* __restrict__ V)
{
  const int lane = threadIdx.x & 31;
  const int wave = threadIdx.x >> 5;
  const int half = lane >> 4;    // selects K-interleave group for A operand
  const int l16  = lane & 15;
  const int m0 = blockIdx.x * 64 + wave * 16;
  const int n0 = blockIdx.y * 64;

  v8f acc[4] = {};

  // A operand: row m = m0 + l16; lane<16 holds K {0..7,16..23}, lane>=16 {8..15,24..31}
  const __bf16* aptr = xb + (size_t)(m0 + l16) * C_EMB + half * 8;

  for (int k0 = 0; k0 < C_EMB; k0 += 32) {
    v8bf alo = *(const v8bf*)(aptr + k0);
    v8bf ahi = *(const v8bf*)(aptr + k0 + 16);
    v16bf A = cat8(alo, ahi);
    const __bf16* bk = wb + (size_t)(k0 + half * 8) * N3 + n0 + l16;  // B col n = n0+l16, K interleaved like A
    __builtin_prefetch(bk + (size_t)32 * N3, 0, 3);         // next K tile, near caches
#pragma unroll
    for (int ns = 0; ns < 4; ++ns) {
      v16bf B;
#pragma unroll
      for (int s = 0; s < 16; ++s) B[s] = bk[(size_t)((s & 7) + (s >> 3) * 16) * N3 + ns * 16];
      acc[ns] = __builtin_amdgcn_wmma_f32_16x16x32_bf16(
          false, A, false, B, (short)0, acc[ns], false, false);
    }
  }

  const float QSCALE = 0.18033688011112042f;  // (1/sqrt(64)) * log2(e)
#pragma unroll
  for (int ns = 0; ns < 4; ++ns) {
    int n   = n0 + ns * 16 + l16;
    int sec = n >> 10;          // 0=Q 1=K 2=V (64-wide tiles never straddle)
    int nin = n & 1023;
    int h = nin >> 6;
    int d = nin & 63;
    float bv = bias[n];
#pragma unroll
    for (int j = 0; j < 8; ++j) {
      // C/D layout: VGPR j -> row m0 + j (+8 for upper lane half), col = l16
      int m = m0 + j + half * 8;
      int b = m >> 11;
      int t = m & (T_SEQ - 1);
      float val = acc[ns][j] + bv;
      if (sec == 0) {
        Q[((size_t)(b * N_HEADS + h) * T_SEQ + t) * D_HEAD + d] = f2bf(val * QSCALE);
      } else if (sec == 1) {
        Kt[((size_t)(b * N_HEADS + h) * D_HEAD + d) * T_SEQ + t] = f2bf(val);
      } else {
        V[((size_t)(b * N_HEADS + h) * T_SEQ + t) * D_HEAD + d] = f2bf(val);
      }
    }
  }
}

// ---------------------------------------------------------------------------
// Flash attention: grid (T/64, H, B), 4 waves/block, each wave owns 16 q-rows.
// Per block: double-buffered 64-key K^T/V tiles staged in LDS with
// GLOBAL_LOAD_ASYNC_TO_LDS_B128 (ASYNCcnt) shared by all 4 waves.
// Per 64-key tile per wave: 8 WMMAs for S = Q*Kt, online softmax (exp2 domain,
// xor-shuffle row reductions), per-wave LDS re-layout of P, 8 WMMAs for O+=P*V.
// ---------------------------------------------------------------------------
__global__ void __launch_bounds__(128) attn_fwd(
    const __bf16* __restrict__ Q, const __bf16* __restrict__ Kt,
    const __bf16* __restrict__ V, float* __restrict__ out)
{
  // K^T tile: row = d (0..63), col = local key.  V tile: row = local key, col = d.
  __shared__ __align__(16) __bf16 KtT[2][64][72];   // 2 x 9216 B
  __shared__ __align__(16) __bf16 Vt[2][64][72];    // 2 x 9216 B
  __shared__ __align__(16) __bf16 Plds[4][16][72];  // per-wave P re-layout

  const int tid  = threadIdx.x;
  const int lane = tid & 31;
  const int wave = tid >> 5;
  const int half = lane >> 4;
  const int l16  = lane & 15;
  const int b = blockIdx.z;
  const int h = blockIdx.y;
  const int qb0 = blockIdx.x * 64;
  const int qb  = qb0 + wave * 16;       // first q row of this wave

  const __bf16* Qbh  = Q  + (size_t)(b * N_HEADS + h) * T_SEQ * D_HEAD;
  const __bf16* Ktbh = Kt + (size_t)(b * N_HEADS + h) * D_HEAD * T_SEQ;
  const __bf16* Vbh  = V  + (size_t)(b * N_HEADS + h) * T_SEQ * D_HEAD;

  // Load Q A-operands once: two K-steps covering d=0..31 and d=32..63
  v16bf qa[2];
  {
    const __bf16* qrow = Qbh + (size_t)(qb + l16) * D_HEAD + half * 8;
#pragma unroll
    for (int ks = 0; ks < 2; ++ks) {
      v8bf lo = *(const v8bf*)(qrow + ks * 32);
      v8bf hi = *(const v8bf*)(qrow + ks * 32 + 16);
      qa[ks] = cat8(lo, hi);
    }
  }

  v8f o[4] = {};
  float rm[8], rl[8];
#pragma unroll
  for (int j = 0; j < 8; ++j) { rm[j] = -1e30f; rl[j] = 0.0f; }

  const unsigned klds0 = (unsigned)(size_t)&KtT[0][0][0];
  const unsigned vlds0 = (unsigned)(size_t)&Vt[0][0][0];

  // Issue the 512 x (16B+16B) async copies for one 64-key tile (8 per thread).
  auto issue_tile = [&](int tile, int buf) {
    const char* kg = (const char*)Ktbh + (size_t)tile * 64 * 2;          // col offset (bytes)
    const char* vg = (const char*)Vbh + (size_t)tile * 64 * (D_HEAD * 2); // row offset (bytes)
    const unsigned kb = klds0 + (unsigned)buf * 64u * 144u;
    const unsigned vb = vlds0 + (unsigned)buf * 64u * 144u;
#pragma unroll
    for (int i = 0; i < 4; ++i) {
      int c   = tid + 128 * i;          // 512 chunks of 16B per tensor
      int row = c >> 3;
      int off = (c & 7) * 16;
      async_cp16(kb + (unsigned)(row * 144 + off),
                 kg + (size_t)row * (T_SEQ * 2) + off);
      async_cp16(vb + (unsigned)(row * 144 + off),
                 vg + (size_t)row * (D_HEAD * 2) + off);
    }
  };

  const int nt = blockIdx.x + 1;   // causal: 64-key tiles 0 .. qb0/64
  issue_tile(0, 0);

  for (int it = 0; it < nt; ++it) {
    const int buf = it & 1;
    if (it + 1 < nt) {
      issue_tile(it + 1, buf ^ 1);                       // overlap DMA with compute
      asm volatile("s_wait_asynccnt 0x8" ::: "memory");  // tile `it` done (next 8 in flight)
    } else {
      asm volatile("s_wait_asynccnt 0x0" ::: "memory");
    }
    __syncthreads();

    const int kt = it * 64;
    const __bf16* kbase = &KtT[buf][0][0];
    const __bf16* vbase = &Vt[buf][0][0];

    // ---- S = Q * K^T  (16 q-rows x 64 keys), B operand from LDS ----
    v8f s[4] = {};
#pragma unroll
    for (int ks = 0; ks < 2; ++ks) {
      const __bf16* kb = kbase + (ks * 32 + half * 8) * 72 + l16;   // B col = key, d interleaved like A
#pragma unroll
      for (int ns = 0; ns < 4; ++ns) {
        v16bf kbv;
#pragma unroll
        for (int e = 0; e < 16; ++e) kbv[e] = kb[((e & 7) + (e >> 3) * 16) * 72 + ns * 16];
        s[ns] = __builtin_amdgcn_wmma_f32_16x16x32_bf16(
            false, qa[ks], false, kbv, (short)0, s[ns], false, false);
      }
    }
    // ---- causal mask (only the diagonal tile needs it) ----
    if (kt + 63 > qb) {
#pragma unroll
      for (int ns = 0; ns < 4; ++ns) {
        int key = kt + ns * 16 + l16;
#pragma unroll
        for (int j = 0; j < 8; ++j) {
          int qr = qb + j + half * 8;
          s[ns][j] = (key > qr) ? -1e30f : (float)s[ns][j];
        }
      }
    }
    // ---- online softmax (exp2 domain; scale folded into Q) ----
#pragma unroll
    for (int j = 0; j < 8; ++j) {
      float mx = fmaxf(fmaxf(s[0][j], s[1][j]), fmaxf(s[2][j], s[3][j]));
      mx = fmaxf(mx, __shfl_xor(mx, 1));
      mx = fmaxf(mx, __shfl_xor(mx, 2));
      mx = fmaxf(mx, __shfl_xor(mx, 4));
      mx = fmaxf(mx, __shfl_xor(mx, 8));   // row max within 16-lane half
      float mnew  = fmaxf(rm[j], mx);
      float alpha = __builtin_amdgcn_exp2f(rm[j] - mnew);
      rm[j] = mnew;
      float rs = 0.0f;
#pragma unroll
      for (int ns = 0; ns < 4; ++ns) {
        float p = __builtin_amdgcn_exp2f(s[ns][j] - mnew);
        s[ns][j] = p;
        rs += p;
      }
      rs += __shfl_xor(rs, 1);
      rs += __shfl_xor(rs, 2);
      rs += __shfl_xor(rs, 4);
      rs += __shfl_xor(rs, 8);
      rl[j] = rl[j] * alpha + rs;
#pragma unroll
      for (int dd = 0; dd < 4; ++dd) o[dd][j] = o[dd][j] * alpha;
    }
    // ---- re-layout P (C/D layout -> A-operand layout) via per-wave LDS ----
#pragma unroll
    for (int ns = 0; ns < 4; ++ns)
#pragma unroll
      for (int j = 0; j < 8; ++j)
        Plds[wave][j + half * 8][ns * 16 + l16] = f2bf(s[ns][j]);

    v16bf pa[2];  // same-wave LDS RAW: DS ops are in-order per wave (DScnt)
#pragma unroll
    for (int ks = 0; ks < 2; ++ks) {
      const __bf16* pr = &Plds[wave][l16][ks * 32 + half * 8];
      v8bf lo = *(const v8bf*)pr;
      v8bf hi = *(const v8bf*)(pr + 16);
      pa[ks] = cat8(lo, hi);
    }
    // ---- O += P * V, B operand from LDS ----
#pragma unroll
    for (int ks = 0; ks < 2; ++ks) {
      const __bf16* vrow = vbase + (ks * 32 + half * 8) * 72 + l16;  // B col = d, key interleaved like A
#pragma unroll
      for (int dd = 0; dd < 4; ++dd) {
        v16bf vbv;
#pragma unroll
        for (int e = 0; e < 16; ++e) vbv[e] = vrow[((e & 7) + (e >> 3) * 16) * 72 + dd * 16];
        o[dd] = __builtin_amdgcn_wmma_f32_16x16x32_bf16(
            false, pa[ks], false, vbv, (short)0, o[dd], false, false);
      }
    }
    __syncthreads();  // protect LDS K/V buffer before the next overwrite
  }

  // ---- normalize and store fp32 output [B, T, C] ----
#pragma unroll
  for (int j = 0; j < 8; ++j) {
    float inv = __builtin_amdgcn_rcpf(rl[j]);
    int t = qb + j + half * 8;
    float* orow = out + (size_t)(b * T_SEQ + t) * C_EMB + h * D_HEAD;
#pragma unroll
    for (int dd = 0; dd < 4; ++dd)
      orow[dd * 16 + l16] = o[dd][j] * inv;
  }
}

// ---------------------------------------------------------------------------
extern "C" void kernel_launch(void* const* d_in, const int* in_sizes, int n_in,
                              void* d_out, int out_size, void* d_ws, size_t ws_size,
                              hipStream_t stream) {
  const float* x    = (const float*)d_in[0];   // [2,2048,1024]
  const float* w    = (const float*)d_in[1];   // [1024,3072]
  const float* bias = (const float*)d_in[2];   // [3072]
  float* out = (float*)d_out;

  char* ws = (char*)d_ws;
  __bf16* xb = (__bf16*)(ws);                        //  8 MB: x bf16 [4096,1024]
  __bf16* wb = (__bf16*)(ws + ((size_t)8  << 20));   //  6 MB: w bf16 [1024,3072]
  __bf16* Qb = (__bf16*)(ws + ((size_t)14 << 20));   //  8 MB: Q [B,H,T,D] (scaled)
  __bf16* Kt = (__bf16*)(ws + ((size_t)22 << 20));   //  8 MB: K^T [B,H,D,T]
  __bf16* Vb = (__bf16*)(ws + ((size_t)30 << 20));   //  8 MB: V [B,H,T,D]

  const int xn4 = (2 * T_SEQ * C_EMB) / 4;
  const int wn4 = (C_EMB * N3) / 4;
  cvt_f32_bf16<<<(xn4 + 255) / 256, 256, 0, stream>>>((const v4f*)x, (v4bf*)xb, xn4);
  cvt_f32_bf16<<<(wn4 + 255) / 256, 256, 0, stream>>>((const v4f*)w, (v4bf*)wb, wn4);

  // M tiles: 4096/64 = 64, N tiles: 3072/64 = 48
  qkv_gemm<<<dim3(64, 48), 128, 0, stream>>>(xb, wb, bias, Qb, Kt, Vb);

  // (T/64, H, B) blocks, 4 waves each -> shared K/V tiles per 64-row q block
  attn_fwd<<<dim3(T_SEQ / 64, N_HEADS, 2), 128, 0, stream>>>(Qb, Kt, Vb, out);
}
